// EdgeConvBlock_53961969107163
// MI455X (gfx1250) — matrix operand order repaired and checked by reading, not compiled, not run on hardware
//
#include <hip/hip_runtime.h>
#include <hip/hip_bf16.h>

typedef __attribute__((ext_vector_type(16))) _Float16 v16h;
typedef __attribute__((ext_vector_type(8)))  float    v8f;

#define BN_EPS 1e-5f
#define H1_STRIDE 72   // padded LDS row stride (halves): 16B-aligned, bank-conflict-free
#define W2_STRIDE 72

// stats layout (floats): [0:64) sum1 [64:128) ss1 [128:192) sum2 [192:256) ss2
//                        [256:320) scale1 [320:384) shift1 [384:448) scale2 [448:512) shift2

__global__ void k_init(float* __restrict__ out, float* __restrict__ cnt,
                       float* __restrict__ stats, int nOut, int nCnt) {
  int i = blockIdx.x * blockDim.x + threadIdx.x;
  int stride = gridDim.x * blockDim.x;
  for (int j = i; j < nOut; j += stride) out[j] = 0.f;
  for (int j = i; j < nCnt; j += stride) cnt[j] = 0.f;
  if (i < 512) stats[i] = 0.f;
}

// u[n] = x[n] @ (W1a - W1b) + b1 ; v[n] = x[n] @ W1b   (W1 = [W1a;W1b], 128x64)
__global__ void k_node_proj(const float* __restrict__ x, const float* __restrict__ W1,
                            const float* __restrict__ b1, float* __restrict__ u,
                            float* __restrict__ v, int nNodes) {
  __shared__ float Am[64][64];
  __shared__ float Bm[64][64];
  for (int idx = threadIdx.x; idx < 4096; idx += 256) {
    int k = idx >> 6, c = idx & 63;
    float wb = W1[(64 + k) * 64 + c];
    Bm[k][c] = wb;
    Am[k][c] = W1[k * 64 + c] - wb;
  }
  __syncthreads();
  int n = blockIdx.x * 4 + (threadIdx.x >> 6);
  int c = threadIdx.x & 63;
  if (n >= nNodes) return;
  const float* xr = x + (size_t)n * 64;
  float au = b1[c], av = 0.f;
#pragma unroll
  for (int k = 0; k < 64; ++k) {
    float xv = xr[k];
    au = fmaf(xv, Am[k][c], au);
    av = fmaf(xv, Bm[k][c], av);
  }
  u[(size_t)n * 64 + c] = au;
  v[(size_t)n * 64 + c] = av;
}

// per-channel sum / sumsq of y1 = u[dst]+v[src]; also edge counts per dst node
__global__ void k_stats1(const int* __restrict__ srcI, const int* __restrict__ dstI,
                         const float* __restrict__ u, const float* __restrict__ v,
                         float* __restrict__ stats, float* __restrict__ cnt, long long E) {
  __shared__ float lsum[64], lss[64];
  if (threadIdx.x < 64) { lsum[threadIdx.x] = 0.f; lss[threadIdx.x] = 0.f; }
  __syncthreads();
  int c = threadIdx.x & 63;
  int sub = threadIdx.x >> 6;
  float s = 0.f, ss = 0.f;
  for (long long e = (long long)blockIdx.x * 4 + sub; e < E; e += (long long)gridDim.x * 4) {
    int d = dstI[e], sn = srcI[e];
    float y = u[(size_t)d * 64 + c] + v[(size_t)sn * 64 + c];
    s += y; ss += y * y;
    if (c == 0) atomicAdd(&cnt[d], 1.f);
  }
  atomicAdd(&lsum[c], s);
  atomicAdd(&lss[c], ss);
  __syncthreads();
  if (threadIdx.x < 64) {
    atomicAdd(&stats[threadIdx.x], lsum[threadIdx.x]);
    atomicAdd(&stats[64 + threadIdx.x], lss[threadIdx.x]);
  }
}

// BN1 scale/shift + pack W2 transposed to f16 [c][k] for B-fragment loads
__global__ void k_finalize1(const float* __restrict__ g1, const float* __restrict__ be1,
                            const float* __restrict__ W2, float* __restrict__ stats,
                            _Float16* __restrict__ w2t, float invE) {
  int c = threadIdx.x;  // 64 threads
  float mu = stats[c] * invE;
  float var = stats[64 + c] * invE - mu * mu;
  float sc = g1[c] * rsqrtf(var + BN_EPS);
  stats[256 + c] = sc;
  stats[320 + c] = be1[c] - mu * sc;
  for (int k = 0; k < 64; ++k) w2t[c * 64 + k] = (_Float16)W2[k * 64 + c];
}

__global__ void k_finalize2(const float* __restrict__ g2, const float* __restrict__ be2,
                            float* __restrict__ stats, float invE) {
  int c = threadIdx.x;
  float mu = stats[128 + c] * invE;
  float var = stats[192 + c] * invE - mu * mu;
  float sc = g2[c] * rsqrtf(var + BN_EPS);
  stats[384 + c] = sc;
  stats[448 + c] = be2[c] - mu * sc;
}

__device__ __forceinline__ v16h ld_frag16(const _Float16* __restrict__ q) {
  v16h f;
#pragma unroll
  for (int i = 0; i < 16; ++i) f[i] = q[i < 8 ? i : i + 8];
  return f;
}

// build 16x64 f16 h1 tile in this wave's LDS slice (per-wave sync only)
__device__ __forceinline__ void tile_fill_h1(const int* __restrict__ srcI,
                                             const int* __restrict__ dstI, long long e0,
                                             const float* __restrict__ u,
                                             const float* __restrict__ v, float sc0, float sc1,
                                             float sh0, float sh1, _Float16* h1, int* dl, int* sl,
                                             int lane) {
  if (lane < 16) { dl[lane] = dstI[e0 + lane]; sl[lane] = srcI[e0 + lane]; }
  __builtin_amdgcn_wave_barrier();
  asm volatile("s_wait_dscnt 0" ::: "memory");
  int c0 = lane * 2;
#pragma unroll
  for (int r = 0; r < 16; ++r) {
    int d = dl[r], s = sl[r];
    float ya = u[(size_t)d * 64 + c0] + v[(size_t)s * 64 + c0];
    float yb = u[(size_t)d * 64 + c0 + 1] + v[(size_t)s * 64 + c0 + 1];
    h1[r * H1_STRIDE + c0]     = (_Float16)fmaxf(fmaf(ya, sc0, sh0), 0.f);
    h1[r * H1_STRIDE + c0 + 1] = (_Float16)fmaxf(fmaf(yb, sc1, sh1), 0.f);
  }
  __builtin_amdgcn_wave_barrier();
  asm volatile("s_wait_dscnt 0" ::: "memory");
}

// y2 tile = h1(16x64) @ W2(64x64); B fragments streamed from LDS (no register residency)
__device__ __forceinline__ void tile_mma(const _Float16* __restrict__ h1,
                                         const _Float16* __restrict__ w2s, v8f acc[4], int lane) {
  int row = lane & 15, hi = lane >> 4, n = lane & 15;
  const _Float16* p = h1 + row * H1_STRIDE + 8 * hi;
  v16h a0, a1;
#pragma unroll
  for (int j = 0; j < 8; ++j) {
    a0[j] = p[j];       a0[j + 8] = p[j + 16];   // K 0..31 slice per A layout
    a1[j] = p[j + 32];  a1[j + 8] = p[j + 48];   // K 32..63
  }
#pragma unroll
  for (int t = 0; t < 4; ++t) {
    const _Float16* q = w2s + (t * 16 + n) * W2_STRIDE + 8 * hi;
    v8f c = {0.f, 0.f, 0.f, 0.f, 0.f, 0.f, 0.f, 0.f};
    c = __builtin_amdgcn_wmma_f32_16x16x32_f16(false, a0, false, ld_frag16(q), (short)0, c, false,
                                               false);
    c = __builtin_amdgcn_wmma_f32_16x16x32_f16(false, a1, false, ld_frag16(q + 32), (short)0, c,
                                               false, false);
    acc[t] = c;
  }
}

__device__ __forceinline__ void load_w2s(const _Float16* __restrict__ w2t,
                                         _Float16* __restrict__ w2s) {
  for (int idx = threadIdx.x; idx < 4096; idx += 256) {
    int c = idx >> 6, k = idx & 63;
    w2s[c * W2_STRIDE + k] = w2t[c * 64 + k];
  }
}

__global__ void __launch_bounds__(256, 2)
k_stats2(const int* __restrict__ srcI, const int* __restrict__ dstI, const float* __restrict__ u,
         const float* __restrict__ v, float* __restrict__ stats,
         const _Float16* __restrict__ w2t, const float* __restrict__ b2, long long nTiles) {
  __shared__ _Float16 h1s[8][16 * H1_STRIDE];
  __shared__ _Float16 w2s[64 * W2_STRIDE];
  __shared__ int dls[8][16], sls[8][16];
  __shared__ float lsum[64], lss[64];
  if (threadIdx.x < 64) { lsum[threadIdx.x] = 0.f; lss[threadIdx.x] = 0.f; }
  load_w2s(w2t, w2s);
  __syncthreads();
  int lane = threadIdx.x & 31, w = threadIdx.x >> 5;
  int n = lane & 15;
  float b2v[4];
#pragma unroll
  for (int t = 0; t < 4; ++t) b2v[t] = b2[t * 16 + n];
  float sc0 = stats[256 + 2 * lane], sc1 = stats[256 + 2 * lane + 1];
  float sh0 = stats[320 + 2 * lane], sh1 = stats[320 + 2 * lane + 1];
  float s[4] = {0, 0, 0, 0}, ss[4] = {0, 0, 0, 0};
  long long wid = (long long)blockIdx.x * 8 + w;
  long long wstride = (long long)gridDim.x * 8;
  for (long long tile = wid; tile < nTiles; tile += wstride) {
    tile_fill_h1(srcI, dstI, tile * 16, u, v, sc0, sc1, sh0, sh1, h1s[w], dls[w], sls[w], lane);
    v8f acc[4];
    tile_mma(h1s[w], w2s, acc, lane);
#pragma unroll
    for (int t = 0; t < 4; ++t)
#pragma unroll
      for (int r = 0; r < 8; ++r) {
        float y = acc[t][r] + b2v[t];
        s[t] += y; ss[t] += y * y;
      }
  }
#pragma unroll
  for (int t = 0; t < 4; ++t) { atomicAdd(&lsum[t * 16 + n], s[t]); atomicAdd(&lss[t * 16 + n], ss[t]); }
  __syncthreads();
  if (threadIdx.x < 64) {
    atomicAdd(&stats[128 + threadIdx.x], lsum[threadIdx.x]);
    atomicAdd(&stats[192 + threadIdx.x], lss[threadIdx.x]);
  }
}

__global__ void __launch_bounds__(256, 2)
k_final(const int* __restrict__ srcI, const int* __restrict__ dstI, const float* __restrict__ u,
        const float* __restrict__ v, const float* __restrict__ stats,
        const _Float16* __restrict__ w2t, const float* __restrict__ b2, float* __restrict__ out,
        long long nTiles) {
  __shared__ _Float16 h1s[8][16 * H1_STRIDE];
  __shared__ _Float16 w2s[64 * W2_STRIDE];
  __shared__ int dls[8][16], sls[8][16];
  load_w2s(w2t, w2s);
  __syncthreads();
  int lane = threadIdx.x & 31, w = threadIdx.x >> 5;
  int n = lane & 15, hi = lane >> 4;
  float b2v[4], sc2[4], sh2[4];
#pragma unroll
  for (int t = 0; t < 4; ++t) {
    b2v[t] = b2[t * 16 + n];
    sc2[t] = stats[384 + t * 16 + n];
    sh2[t] = stats[448 + t * 16 + n];
  }
  float sc0 = stats[256 + 2 * lane], sc1 = stats[256 + 2 * lane + 1];
  float sh0 = stats[320 + 2 * lane], sh1 = stats[320 + 2 * lane + 1];
  long long wid = (long long)blockIdx.x * 8 + w;
  long long wstride = (long long)gridDim.x * 8;
  for (long long tile = wid; tile < nTiles; tile += wstride) {
    tile_fill_h1(srcI, dstI, tile * 16, u, v, sc0, sc1, sh0, sh1, h1s[w], dls[w], sls[w], lane);
    v8f acc[4];
    tile_mma(h1s[w], w2s, acc, lane);
#pragma unroll
    for (int t = 0; t < 4; ++t)
#pragma unroll
      for (int r = 0; r < 8; ++r) {
        float y = acc[t][r] + b2v[t];
        float z = fmaxf(fmaf(y, sc2[t], sh2[t]), 0.f);
        int node = dls[w][r + 8 * hi];
        atomicAdd(&out[(size_t)node * 64 + t * 16 + n], z);
      }
  }
}

// scalar fallbacks for a ragged tail (E % 16 != 0) — not hit for E = 1.6M
__global__ void k_tail_stats2(const int* srcI, const int* dstI, const float* u, const float* v,
                              const float* stats, const float* W2, const float* b2, float* gstats,
                              long long eStart, long long E) {
  long long e = eStart + blockIdx.x;
  if (e >= E) return;
  int c = threadIdx.x;
  int d = dstI[e], sn = srcI[e];
  float y = b2[c];
  for (int k = 0; k < 64; ++k) {
    float h = fmaxf(fmaf(u[(size_t)d * 64 + k] + v[(size_t)sn * 64 + k], stats[256 + k],
                         stats[320 + k]), 0.f);
    y = fmaf(h, W2[k * 64 + c], y);
  }
  atomicAdd(&gstats[128 + c], y);
  atomicAdd(&gstats[192 + c], y * y);
}

__global__ void k_tail_final(const int* srcI, const int* dstI, const float* u, const float* v,
                             const float* stats, const float* W2, const float* b2, float* out,
                             long long eStart, long long E) {
  long long e = eStart + blockIdx.x;
  if (e >= E) return;
  int c = threadIdx.x;
  int d = dstI[e], sn = srcI[e];
  float y = b2[c];
  for (int k = 0; k < 64; ++k) {
    float h = fmaxf(fmaf(u[(size_t)d * 64 + k] + v[(size_t)sn * 64 + k], stats[256 + k],
                         stats[320 + k]), 0.f);
    y = fmaf(h, W2[k * 64 + c], y);
  }
  float z = fmaxf(fmaf(y, stats[384 + c], stats[448 + c]), 0.f);
  atomicAdd(&out[(size_t)d * 64 + c], z);
}

__global__ void k_div(float* __restrict__ out, const float* __restrict__ cnt, int nNodes) {
  int i = blockIdx.x * blockDim.x + threadIdx.x;
  if (i >= nNodes * 64) return;
  out[i] /= fmaxf(cnt[i >> 6], 1.f);
}

extern "C" void kernel_launch(void* const* d_in, const int* in_sizes, int n_in, void* d_out,
                              int out_size, void* d_ws, size_t ws_size, hipStream_t stream) {
  const float* x   = (const float*)d_in[0];
  const int*   ei  = (const int*)d_in[1];  // int32 on device (JAX w/o x64)
  const float* W1  = (const float*)d_in[2];
  const float* b1  = (const float*)d_in[3];
  const float* g1  = (const float*)d_in[4];
  const float* be1 = (const float*)d_in[5];
  const float* W2  = (const float*)d_in[6];
  const float* b2  = (const float*)d_in[7];
  const float* g2  = (const float*)d_in[8];
  const float* be2 = (const float*)d_in[9];
  float* out = (float*)d_out;

  int nNodes = in_sizes[0] / 64;
  long long E = in_sizes[1] / 2;
  const int* srcI = ei;       // edge_index[0]
  const int* dstI = ei + E;   // edge_index[1]

  float* u = (float*)d_ws;
  float* v = u + (size_t)nNodes * 64;
  float* stats = v + (size_t)nNodes * 64;
  float* cnt = stats + 512;
  _Float16* w2t = (_Float16*)(cnt + nNodes);

  float invE = 1.0f / (float)E;
  long long nTiles = E >> 4;
  long long eTail = nTiles << 4;
  int rem = (int)(E - eTail);

  k_init<<<2048, 256, 0, stream>>>(out, cnt, stats, nNodes * 64, nNodes);
  k_node_proj<<<(nNodes + 3) / 4, 256, 0, stream>>>(x, W1, b1, u, v, nNodes);
  k_stats1<<<2048, 256, 0, stream>>>(srcI, dstI, u, v, stats, cnt, E);
  k_finalize1<<<1, 64, 0, stream>>>(g1, be1, W2, stats, w2t, invE);
  k_stats2<<<1024, 256, 0, stream>>>(srcI, dstI, u, v, stats, w2t, b2, nTiles);
  if (rem) k_tail_stats2<<<rem, 64, 0, stream>>>(srcI, dstI, u, v, stats, W2, b2, stats, eTail, E);
  k_finalize2<<<1, 64, 0, stream>>>(g2, be2, stats, invE);
  k_final<<<1024, 256, 0, stream>>>(srcI, dstI, u, v, stats, w2t, b2, out, nTiles);
  if (rem) k_tail_final<<<rem, 64, 0, stream>>>(srcI, dstI, u, v, stats, W2, b2, out, eTail, E);
  k_div<<<(nNodes * 64 + 255) / 256, 256, 0, stream>>>(out, cnt, nNodes);
}
